// AnatomicalScanMamba_52278341927353
// MI455X (gfx1250) — compile-verified, matrix operand order checked by reading, not run on hardware
//
#include <hip/hip_runtime.h>
#include <hip/hip_bf16.h>
#include <math.h>

// ---------------- model dims ----------------
#define D_MODEL   128
#define D_STATE   16
#define D_CONV    4
#define D_INNER   256
#define DT_RANK   8
#define NJ        24
#define NPATH     5
#define MAXL      6
#define NB        512
#define NSEQ      (NB * NPATH)          // 2560
#define NROW      (NSEQ * MAXL)         // 15360 position rows
#define SEQ_PER_WG 8
#define ROWS      (SEQ_PER_WG * MAXL)   // 48 rows / workgroup (3 WMMA row tiles)

typedef __attribute__((ext_vector_type(16))) _Float16 v16h;
typedef __attribute__((ext_vector_type(8)))  _Float16 v8h;
typedef __attribute__((ext_vector_type(8)))  float    v8f;

// ---- WMMA tile helpers (CDNA5 16-bit A layout; B uses symmetric layout) ----
// lane<16 : row = lane,     K elems {0..7, 16..23}
// lane>=16: row = lane-16,  K elems {8..15, 24..31}
__device__ static inline v16h load_tile(const _Float16* p, int ld) {
  int lane = threadIdx.x & 31;
  const _Float16* q = p + (lane & 15) * ld + ((lane & 16) ? 8 : 0);
  v16h v;
#pragma unroll
  for (int i = 0; i < 8; ++i) v[i] = q[i];
#pragma unroll
  for (int i = 0; i < 8; ++i) v[8 + i] = q[16 + i];
  return v;
}

// f32 16x16 C/D layout: lane<16 -> N=lane, M=r ; lane>=16 -> N=lane-16, M=8+r
__device__ static inline void store_tile_f16(_Float16* p, int ld, v8f c) {
  int lane = threadIdx.x & 31;
  _Float16* q = p + (lane & 15) + ((lane & 16) ? 8 : 0) * ld;
#pragma unroll
  for (int r = 0; r < 8; ++r) q[r * ld] = (_Float16)c[r];
}
__device__ static inline void store_tile_f32(float* p, int ld, v8f c) {
  int lane = threadIdx.x & 31;
  float* q = p + (lane & 15) + ((lane & 16) ? 8 : 0) * ld;
#pragma unroll
  for (int r = 0; r < 8; ++r) q[r * ld] = c[r];
}

__device__ static inline float siluf(float v) { return v / (1.f + expf(-v)); }

// ---- CDNA5 async global->LDS staging (ASYNCcnt-tracked, no VGPR round trip) ----
// LDS address comes from the vdst VGPR; low 32 bits of a flat shared pointer are the
// in-aperture LDS byte offset (ISA 10.2 aperture mapping).
__device__ static inline void async_g2l_b128(unsigned lds_byte, const void* gptr) {
  asm volatile("global_load_async_to_lds_b128 %0, %1, off"
               :: "v"(lds_byte), "v"(gptr) : "memory");
}
__device__ static inline void wait_async0() {
  asm volatile("s_wait_asynccnt 0x0" ::: "memory");
}

// ---------------- kernel 1: weight packing ----------------
__global__ void k_prep(const float* __restrict__ WinF, const float* __restrict__ WinB,
                       const float* __restrict__ WxF,  const float* __restrict__ WxB,
                       const float* __restrict__ AlF,  const float* __restrict__ AlB,
                       const float* __restrict__ fusion_W,
                       _Float16* __restrict__ WinHF, _Float16* __restrict__ WinHB,
                       _Float16* __restrict__ WxHF,  _Float16* __restrict__ WxHB,
                       float* __restrict__ AF, float* __restrict__ AB,
                       _Float16* __restrict__ RsumH) {
  int i = blockIdx.x * 256 + threadIdx.x;
  if (i < 2 * D_INNER * D_MODEL) { WinHF[i] = (_Float16)WinF[i]; WinHB[i] = (_Float16)WinB[i]; }
  if (i < (DT_RANK + 2 * D_STATE) * D_INNER) { WxHF[i] = (_Float16)WxF[i]; WxHB[i] = (_Float16)WxB[i]; }
  if (i < D_INNER * D_STATE) { AF[i] = -expf(AlF[i]); AB[i] = -expf(AlB[i]); }
  if (i < D_MODEL * D_MODEL) {
    int m = i >> 7, k = i & 127;
    RsumH[i] = (_Float16)(fusion_W[m * 256 + k] + fusion_W[m * 256 + 128 + k]);
  }
}

// Wcomb_d[m,e] = sum_dd fusion_W[m, off+dd] * W_out_d[dd,e]   (128 x 256)
__global__ void k_wcomb(const float* __restrict__ fusion_W,
                        const float* __restrict__ WoutF, const float* __restrict__ WoutB,
                        _Float16* __restrict__ WcF, _Float16* __restrict__ WcB) {
  int i = blockIdx.x * 256 + threadIdx.x;   // 0..32767
  int m = i >> 8, e = i & 255;
  float af = 0.f, ab = 0.f;
  for (int d = 0; d < 128; ++d) {
    af += fusion_W[m * 256 + d]       * WoutF[d * 256 + e];
    ab += fusion_W[m * 256 + 128 + d] * WoutB[d * 256 + e];
  }
  WcF[i] = (_Float16)af; WcB[i] = (_Float16)ab;
}

// ---------------- kernel 2: gather + dual LayerNorm ----------------
__global__ void k_gather_ln(const float* __restrict__ x, const int* __restrict__ path,
                            const float* __restrict__ gF, const float* __restrict__ bF,
                            const float* __restrict__ gB, const float* __restrict__ bB,
                            _Float16* __restrict__ seqh,
                            _Float16* __restrict__ lnF, _Float16* __restrict__ lnB) {
  int row = blockIdx.x, t = threadIdx.x;      // row in [0,15360), t in [0,128)
  int n = row / MAXL, l = row - n * MAXL;
  int b = n / NPATH, p = n - b * NPATH;
  int j = path[p * MAXL + l];
  float v = (j >= 0) ? x[((size_t)b * NJ + j) * D_MODEL + t] : 0.f;
  __shared__ float red[128];
  red[t] = v; __syncthreads();
#pragma unroll
  for (int s = 64; s > 0; s >>= 1) { if (t < s) red[t] += red[t + s]; __syncthreads(); }
  float mu = red[0] * (1.f / 128.f);
  __syncthreads();
  float d = v - mu;
  red[t] = d * d; __syncthreads();
#pragma unroll
  for (int s = 64; s > 0; s >>= 1) { if (t < s) red[t] += red[t + s]; __syncthreads(); }
  float inv = rsqrtf(red[0] * (1.f / 128.f) + 1e-5f);
  float xn = d * inv;
  size_t o = (size_t)row * D_MODEL + t;
  seqh[o] = (_Float16)v;
  lnF[o] = (_Float16)(xn * gF[t] + bF[t]);
  lnB[o] = (_Float16)(xn * gB[t] + bB[t]);
}

// ---------------- kernel 3: fused Mamba mixer (one direction) ----------------
// Per WG: 8 sequences (48 rows). LDS: [0,12288)=ln tile (later dt/B/C f32),
// [12288,61440)=xz f16 (48x512). contrib = y_gated @ Wcomb^T  (folded fusion).
__global__ void __launch_bounds__(256) k_mixer(
    const _Float16* __restrict__ lnh,     // (15360,128) f16
    const _Float16* __restrict__ WinH,    // (512,128)   f16
    const _Float16* __restrict__ WxH,     // (40,256)    f16
    const _Float16* __restrict__ WcombH,  // (128,256)   f16
    const float* __restrict__ conv_w,     // (256,4)
    const float* __restrict__ conv_b,     // (256)
    const float* __restrict__ W_dt,       // (256,8)
    const float* __restrict__ b_dt,       // (256)
    const float* __restrict__ Aneg,       // (256,16)  = -exp(A_log)
    const float* __restrict__ Dsk,        // (256)
    float* __restrict__ contrib,          // (15360,128) f32
    int dir) {
  __shared__ __align__(16) char smem[61440];
  _Float16* s_ln  = (_Float16*)smem;            // 48x128 f16
  float*    s_dbc = (float*)smem;               // 48x40  f32 (aliases s_ln after GEMM1)
  _Float16* s_xz  = (_Float16*)(smem + 12288);  // 48x512 f16

  const int tid  = threadIdx.x;
  const int wave = tid >> 5;
  const int row0 = blockIdx.x * ROWS;

  // stage 0: async-stage LN tile into LDS (768 x 16B chunks, 3 per thread)
  {
    const char* g = (const char*)(lnh + (size_t)row0 * D_MODEL);
    unsigned lbase = (unsigned)(size_t)s_ln;
    for (int i = tid; i < ROWS * D_MODEL * 2 / 16; i += 256)
      async_g2l_b128(lbase + i * 16, g + (size_t)i * 16);
    wait_async0();
  }
  __syncthreads();

  // stage 1: xz = ln @ Win^T  (48x128 @ 128x512) via WMMA, f16 result in LDS
  for (int t = wave; t < 3 * 32; t += 8) {
    int rt = t >> 5, ct = t & 31;
    v8f acc = {};
#pragma unroll
    for (int kk = 0; kk < 4; ++kk) {
      v16h a = load_tile(s_ln + (rt * 16) * 128 + kk * 32, 128);
      v16h b = load_tile(WinH + (ct * 16) * 128 + kk * 32, 128);
      acc = __builtin_amdgcn_wmma_f32_16x16x32_f16(false, a, false, b, (short)0, acc,
                                                   false, false);
    }
    store_tile_f16(s_xz + (rt * 16) * 512 + ct * 16, 512, acc);
  }
  __syncthreads();

  // stage 2: depthwise causal conv + SiLU, direction-aware, in place on cols [0,256)
  {
    float pre[SEQ_PER_WG][MAXL];
#pragma unroll
    for (int k = 0; k < 8; ++k) {
      int c = tid + 256 * k, s = c >> 8, e = c & 255;
#pragma unroll
      for (int l = 0; l < MAXL; ++l) pre[k][l] = (float)s_xz[(s * MAXL + l) * 512 + e];
    }
    __syncthreads();
#pragma unroll
    for (int k = 0; k < 8; ++k) {
      int c = tid + 256 * k, s = c >> 8, e = c & 255;
      float w[4] = {conv_w[e * 4 + 0], conv_w[e * 4 + 1], conv_w[e * 4 + 2], conv_w[e * 4 + 3]};
      float cb = conv_b[e];
#pragma unroll
      for (int l = 0; l < MAXL; ++l) {
        float acc = cb;
#pragma unroll
        for (int t2 = 0; t2 < 4; ++t2) {
          int lp = (dir == 0) ? (l - 3 + t2) : (l + 3 - t2);
          if (lp >= 0 && lp < MAXL) acc += w[t2] * pre[k][lp];
        }
        s_xz[(s * MAXL + l) * 512 + e] = (_Float16)siluf(acc);
      }
    }
  }
  __syncthreads();

  // stage 3: xdbl = xh @ Wx^T -> dt_raw(8)|B(16)|C(16), f32 (48x40), vectorized VALU
  for (int i = tid; i < ROWS * 40; i += 256) {
    int r = i / 40, f = i - r * 40;
    const v8h* xr = (const v8h*)(s_xz + r * 512);   // 32 chunks of 8 f16 (b128 loads)
    const v8h* wr = (const v8h*)(WxH + f * 256);
    float acc = 0.f;
#pragma unroll
    for (int k = 0; k < 32; ++k) {
      v8h a = xr[k], b = wr[k];
#pragma unroll
      for (int u = 0; u < 8; ++u) acc += (float)a[u] * (float)b[u];
    }
    s_dbc[i] = acc;
  }
  __syncthreads();

  // stage 4: selective scan; overwrite z slot with (y + D*xh)*silu(z) as f16
  {
    int e = tid;  // one channel per thread
    float wdt[DT_RANK];
#pragma unroll
    for (int q = 0; q < DT_RANK; ++q) wdt[q] = W_dt[e * DT_RANK + q];
    float bd = b_dt[e], Ds = Dsk[e];
    float Ar[D_STATE];
#pragma unroll
    for (int j = 0; j < D_STATE; ++j) Ar[j] = Aneg[e * D_STATE + j];
    for (int s = 0; s < SEQ_PER_WG; ++s) {
      float h[D_STATE];
#pragma unroll
      for (int j = 0; j < D_STATE; ++j) h[j] = 0.f;
      for (int st = 0; st < MAXL; ++st) {
        int l = (dir == 0) ? st : (MAXL - 1 - st);
        int r = s * MAXL + l;
        const float* db = s_dbc + r * 40;
        float draw = bd;
#pragma unroll
        for (int q = 0; q < DT_RANK; ++q) draw += db[q] * wdt[q];
        float dt = (draw > 20.f) ? draw : log1pf(expf(draw));   // softplus
        float xhv = (float)s_xz[r * 512 + e];
        float y = 0.f;
#pragma unroll
        for (int j = 0; j < D_STATE; ++j) {
          float a  = expf(dt * Ar[j]);
          float bx = dt * db[8 + j] * xhv;
          h[j] = a * h[j] + bx;
          y += h[j] * db[24 + j];
        }
        float z = (float)s_xz[r * 512 + 256 + e];
        s_xz[r * 512 + 256 + e] = (_Float16)((y + Ds * xhv) * siluf(z));
      }
    }
  }
  __syncthreads();

  // stage 5: contrib = y_gated @ Wcomb^T  (48x256 @ 256x128) via WMMA, f32 to HBM
  for (int t = wave; t < 3 * 8; t += 8) {
    int rt = t >> 3, ct = t & 7;
    v8f acc = {};
#pragma unroll
    for (int kk = 0; kk < 8; ++kk) {
      v16h a = load_tile(s_xz + (rt * 16) * 512 + 256 + kk * 32, 512);
      v16h b = load_tile(WcombH + (ct * 16) * 256 + kk * 32, 256);
      acc = __builtin_amdgcn_wmma_f32_16x16x32_f16(false, a, false, b, (short)0, acc,
                                                   false, false);
    }
    store_tile_f32(contrib + (size_t)(row0 + rt * 16) * D_MODEL + ct * 16, D_MODEL, acc);
  }
}

// ---------------- kernel 4: residual-through-fusion  seq @ Rsum^T ----------------
__global__ void __launch_bounds__(256) k_resfuse(const _Float16* __restrict__ seqh,
                                                 const _Float16* __restrict__ RsumH,
                                                 float* __restrict__ contribR) {
  __shared__ __align__(16) _Float16 s_a[ROWS * D_MODEL];
  const int tid = threadIdx.x, wave = tid >> 5;
  const int row0 = blockIdx.x * ROWS;
  {
    const char* g = (const char*)(seqh + (size_t)row0 * D_MODEL);
    unsigned lbase = (unsigned)(size_t)&s_a[0];
    for (int i = tid; i < ROWS * D_MODEL * 2 / 16; i += 256)
      async_g2l_b128(lbase + i * 16, g + (size_t)i * 16);
    wait_async0();
  }
  __syncthreads();
  for (int t = wave; t < 3 * 8; t += 8) {
    int rt = t >> 3, ct = t & 7;
    v8f acc = {};
#pragma unroll
    for (int kk = 0; kk < 4; ++kk) {
      v16h a = load_tile(s_a + (rt * 16) * 128 + kk * 32, 128);
      v16h b = load_tile(RsumH + (ct * 16) * 128 + kk * 32, 128);
      acc = __builtin_amdgcn_wmma_f32_16x16x32_f16(false, a, false, b, (short)0, acc,
                                                   false, false);
    }
    store_tile_f32(contribR + (size_t)(row0 + rt * 16) * D_MODEL + ct * 16, D_MODEL, acc);
  }
}

// ---------------- kernel 5: scatter back + residual + bias ----------------
__global__ void k_scatter(const float* __restrict__ x,
                          const float* __restrict__ cF, const float* __restrict__ cB,
                          const float* __restrict__ cR, const float* __restrict__ fb,
                          const int* __restrict__ path, float* __restrict__ out) {
  int row = blockIdx.x, t = threadIdx.x;
  int n = row / MAXL, l = row - n * MAXL;
  int b = n / NPATH, p = n - b * NPATH;
  int j = path[p * MAXL + l];
  if (j < 0) return;                       // padded slot -> dropped (paths cover all joints)
  size_t o = ((size_t)b * NJ + j) * D_MODEL + t;
  size_t r = (size_t)row * D_MODEL + t;
  out[o] = x[o] + cF[r] + cB[r] + cR[r] + fb[t];
}

// ---------------- launcher ----------------
extern "C" void kernel_launch(void* const* d_in, const int* in_sizes, int n_in,
                              void* d_out, int out_size, void* d_ws, size_t ws_size,
                              hipStream_t stream) {
  (void)in_sizes; (void)n_in; (void)out_size; (void)ws_size;
  // dict order, lists flattened:
  // 0:x  1..11:f_params  12..22:b_params  23:fusion_W 24:fusion_b 25:path_indices
  // params: 0 ln_g, 1 ln_b, 2 W_in, 3 conv_w, 4 conv_b, 5 W_x, 6 W_dt, 7 b_dt,
  //         8 A_log, 9 Dskip, 10 W_out
  const float* x = (const float*)d_in[0];
  auto F  = [&](int i) { return (const float*)d_in[1 + i]; };
  auto Bp = [&](int i) { return (const float*)d_in[12 + i]; };
  const float* fusion_W = (const float*)d_in[23];
  const float* fusion_b = (const float*)d_in[24];
  const int*   path     = (const int*)d_in[25];

  char* ws = (char*)d_ws;
  size_t off = 0;
  auto carve = [&](size_t bytes) {
    void* p = ws + off;
    off = (off + bytes + 255) & ~(size_t)255;
    return p;
  };
  _Float16* seqh  = (_Float16*)carve((size_t)NROW * D_MODEL * 2);
  _Float16* lnF   = (_Float16*)carve((size_t)NROW * D_MODEL * 2);
  _Float16* lnB   = (_Float16*)carve((size_t)NROW * D_MODEL * 2);
  _Float16* WinHF = (_Float16*)carve(2 * D_INNER * D_MODEL * 2);
  _Float16* WinHB = (_Float16*)carve(2 * D_INNER * D_MODEL * 2);
  _Float16* WxHF  = (_Float16*)carve(40 * D_INNER * 2);
  _Float16* WxHB  = (_Float16*)carve(40 * D_INNER * 2);
  _Float16* WcF   = (_Float16*)carve(D_MODEL * D_INNER * 2);
  _Float16* WcB   = (_Float16*)carve(D_MODEL * D_INNER * 2);
  float*    AF    = (float*)carve(D_INNER * D_STATE * 4);
  float*    AB    = (float*)carve(D_INNER * D_STATE * 4);
  _Float16* RsumH = (_Float16*)carve(D_MODEL * D_MODEL * 2);
  float*    cF    = (float*)carve((size_t)NROW * D_MODEL * 4);
  float*    cB    = (float*)carve((size_t)NROW * D_MODEL * 4);
  float*    cR    = (float*)carve((size_t)NROW * D_MODEL * 4);

  k_prep<<<256, 256, 0, stream>>>(F(2), Bp(2), F(5), Bp(5), F(8), Bp(8), fusion_W,
                                  WinHF, WinHB, WxHF, WxHB, AF, AB, RsumH);
  k_wcomb<<<128, 256, 0, stream>>>(fusion_W, F(10), Bp(10), WcF, WcB);
  k_gather_ln<<<NROW, 128, 0, stream>>>(x, path, F(0), F(1), Bp(0), Bp(1), seqh, lnF, lnB);
  k_mixer<<<NSEQ / SEQ_PER_WG, 256, 0, stream>>>(lnF, WinHF, WxHF, WcF, F(3), F(4),
                                                 F(6), F(7), AF, F(9), cF, 0);
  k_mixer<<<NSEQ / SEQ_PER_WG, 256, 0, stream>>>(lnB, WinHB, WxHB, WcB, Bp(3), Bp(4),
                                                 Bp(6), Bp(7), AB, Bp(9), cB, 1);
  k_resfuse<<<NSEQ / SEQ_PER_WG, 256, 0, stream>>>(seqh, RsumH, cR);
  k_scatter<<<NROW, 128, 0, stream>>>(x, cF, cB, cR, fusion_b, path, (float*)d_out);
}
